// SpeAttention_55216099557745
// MI455X (gfx1250) — matrix operand order repaired and checked by reading, not compiled, not run on hardware
//
#include <hip/hip_runtime.h>
#include <hip/hip_bf16.h>
#include <math.h>

typedef __attribute__((ext_vector_type(16))) _Float16 v16h;
typedef __attribute__((ext_vector_type(8)))  float    v8f;

#define LSEQ 128   // sequence length (= C of input)
#define DM   32    // d_model
#define DSN  8     // d_state
#define NBATCH 16

// ---------------------------------------------------------------------------
// helpers
// ---------------------------------------------------------------------------
__device__ __forceinline__ float sigmoidf_(float x) { return 1.0f / (1.0f + __expf(-x)); }
__device__ __forceinline__ float siluf_(float x)    { return x * sigmoidf_(x); }
__device__ __forceinline__ float softplusf_(float x){ return (x > 20.0f) ? x : log1pf(__expf(x)); }

// A fragment: 16x32 f16 (M x K), row-major source with given stride.
// Per CDNA5 ISA 7.12.2: lane = m%16 + 16*half; VGPR j: k0 = (j<4 ? 2j : 16+2(j-4)) + 8*half
__device__ __forceinline__ v16h load_a16(const _Float16* base, int m0, int stride, int lane) {
  int m  = m0 + (lane & 15);
  int hf = lane >> 4;
  v16h a;
#pragma unroll
  for (int j = 0; j < 8; ++j) {
    int k0 = ((j < 4) ? (2 * j) : (16 + 2 * (j - 4))) + 8 * hf;
    a[2 * j]     = base[m * stride + k0];
    a[2 * j + 1] = base[m * stride + k0 + 1];
  }
  return a;
}

// B fragment: 32x16 f16 (K x N).  B[k][n] = W[n][k], W row-major with given stride.
// lane = n%16 + 16*half; VGPR j holds k = (j<4 ? 2j : 16+2(j-4)) + 8*half (and +1).
__device__ __forceinline__ v16h load_b16(const _Float16* w, int n0, int stride, int lane) {
  int n  = n0 + (lane & 15);
  int hf = lane >> 4;
  v16h b;
#pragma unroll
  for (int j = 0; j < 8; ++j) {
    int k = ((j < 4) ? (2 * j) : (16 + 2 * (j - 4))) + 8 * hf;
    b[2 * j]     = w[n * stride + k];
    b[2 * j + 1] = w[n * stride + k + 1];
  }
  return b;
}

// ---------------------------------------------------------------------------
// Kernel 1: max-pool partials. grid = B*C*4 blocks, 256 threads.
// Each block reduces 16384 contiguous floats (1/4 of a 256x256 plane).
// This kernel carries ~537 MB of HBM traffic -> the only roofline-relevant one.
// ---------------------------------------------------------------------------
__global__ void __launch_bounds__(256) pool_partial(const float* __restrict__ x,
                                                    float* __restrict__ part) {
  int blk   = blockIdx.x;
  int seg   = blk & 3;
  int plane = blk >> 2;                       // b*128 + c
  const float4* p4 = (const float4*)(x + (size_t)plane * 65536 + (size_t)seg * 16384);
  int t = threadIdx.x;
  float m = -INFINITY;
#pragma unroll
  for (int i = 0; i < 16; ++i) {              // 16 * float4 = 64 floats / thread
    float4 v = p4[t + i * 256];
    m = fmaxf(m, fmaxf(fmaxf(v.x, v.y), fmaxf(v.z, v.w)));
  }
  __shared__ float red[256];
  red[t] = m;
  __syncthreads();
  for (int s = 128; s > 0; s >>= 1) {
    if (t < s) red[t] = fmaxf(red[t], red[t + s]);
    __syncthreads();
  }
  if (t == 0) part[blk] = red[0];
}

// Kernel 2: fold 4 partials per (b,c) -> pooled[2048]
__global__ void __launch_bounds__(256) pool_final(const float* __restrict__ part,
                                                  float* __restrict__ pooled) {
  int i = blockIdx.x * 256 + threadIdx.x;
  if (i < NBATCH * LSEQ) {
    float m = part[i * 4];
    m = fmaxf(m, part[i * 4 + 1]);
    m = fmaxf(m, part[i * 4 + 2]);
    m = fmaxf(m, part[i * 4 + 3]);
    pooled[i] = m;
  }
}

// ---------------------------------------------------------------------------
// Kernel 3: fully fused per-batch bidirectional Mamba. grid = 16, block = 256.
// ---------------------------------------------------------------------------
__global__ void __launch_bounds__(256)
mamba_fused(const float* __restrict__ pooled,
            const float* __restrict__ w1w,  const float* __restrict__ w1b,
            const float* __restrict__ lng,  const float* __restrict__ lnb,
            const float* __restrict__ ipw,
            const float* __restrict__ cwf,  const float* __restrict__ cbf,
            const float* __restrict__ cwb,  const float* __restrict__ cbb,
            const float* __restrict__ xpwf, const float* __restrict__ xpwb,
            const float* __restrict__ dpwf, const float* __restrict__ dpbf,
            const float* __restrict__ dpwb, const float* __restrict__ dpbb,
            const float* __restrict__ alogf,const float* __restrict__ alogb,
            const float* __restrict__ Dskf, const float* __restrict__ Dskb,
            const float* __restrict__ rmsw, const float* __restrict__ opw,
            const float* __restrict__ w2w,  const float* __restrict__ w2b,
            float* __restrict__ out) {
  // ---- LDS carve (58368 B total, regions reused across phases) ----
  __shared__ __align__(16) char smem[58368];
  float*    s_dt   = (float*)(smem + 0);          // [2][128]   dt projections
  float*    s_yacc = (float*)(smem + 1024);       // [32][128]  (y_f+y_b)/2 accumulator
  _Float16* s_xzh  = (_Float16*)(smem + 17408);   // [64][128]  xz (f16)
  char*     regC   = smem + 33792;                // 16 KB multi-use
  char*     regD   = smem + 50176;                //  8 KB multi-use
  float*    s_xc   = (float*)regC;                // [32][128]  conv+silu output
  _Float16* s_hn   = (_Float16*)regC;             // [128][32]  LN output (pre-in_proj)
  _Float16* s_w    = (_Float16*)(regC + 8192);    // [64][32]   in_proj weights f16
  float*    s_proj = (float*)regC;                // [128][32]  out_proj result (post-scan)
  float*    s_B    = (float*)regD;                // [8][128]   B(t)
  float*    s_C    = (float*)(regD + 4096);       // [8][128]   C(t)
  _Float16* s_o16  = (_Float16*)regD;             // [128][32]  RMS-normed out (post-scan)
  _Float16* s_wo   = (_Float16*)(smem + 17408);   // [32][32]   out_proj weights (reuses xzh)

  const int t   = threadIdx.x;
  const int bid = blockIdx.x;
  const int lane = t & 31;
  const int wv   = t >> 5;

  // ---- Phase 1: weights->f16, zero yacc, h = pooled*w1+b, LayerNorm ----
  for (int idx = t; idx < 2 * DM * DM; idx += 256) s_w[idx] = (_Float16)ipw[idx];
  for (int idx = t; idx < DM * LSEQ; idx += 256)   s_yacc[idx] = 0.0f;
  if (t < LSEQ) {
    const int l = t;
    const float p = pooled[bid * LSEQ + l];
    float hv[DM];
    float mu = 0.0f;
#pragma unroll
    for (int dd = 0; dd < DM; ++dd) { hv[dd] = p * w1w[dd] + w1b[dd]; mu += hv[dd]; }
    mu *= (1.0f / DM);
    float var = 0.0f;
#pragma unroll
    for (int dd = 0; dd < DM; ++dd) { float d = hv[dd] - mu; var += d * d; }
    var *= (1.0f / DM);
    const float inv = rsqrtf(var + 1e-5f);
#pragma unroll
    for (int dd = 0; dd < DM; ++dd)
      s_hn[l * DM + dd] = (_Float16)((hv[dd] - mu) * inv * lng[dd] + lnb[dd]);
  }
  __syncthreads();

  // ---- Phase 2: in_proj GEMM (128x32)x(32x64) via WMMA, wave w owns 16 rows ----
  {
    const int m0 = wv * 16;
    v16h a = load_a16(s_hn, m0, DM, lane);
#pragma unroll
    for (int nt = 0; nt < 4; ++nt) {
      v16h bf = load_b16(s_w, nt * 16, DM, lane);
      v8f c = {};
      c = __builtin_amdgcn_wmma_f32_16x16x32_f16(false, a, false, bf, (short)0, c, false, false);
      const int n  = nt * 16 + (lane & 15);
      const int hf = lane >> 4;
#pragma unroll
      for (int j = 0; j < 8; ++j)
        s_xzh[n * LSEQ + (m0 + j + 8 * hf)] = (_Float16)c[j];   // xz[e][l]
    }
  }
  __syncthreads();

  // ---- Phase 3: two directional scans ----
  for (int dir = 0; dir < 2; ++dir) {
    const float* cw  = dir ? cwb  : cwf;
    const float* cb  = dir ? cbb  : cbf;
    const float* xpw = dir ? xpwb : xpwf;
    const float* dpw = dir ? dpwb : dpwf;
    const float* dpb = dir ? dpbb : dpbf;
    const float* alg = dir ? alogb : alogf;
    const float* Dsk = dir ? Dskb  : Dskf;

    // causal depthwise conv (k=4, left pad 3) on (optionally reversed) xi + SiLU
    for (int idx = t; idx < DM * LSEQ; idx += 256) {
      const int dd = idx >> 7, j = idx & 127;
      float acc = 0.0f;
#pragma unroll
      for (int tap = 0; tap < 4; ++tap) {
        const int jj = j + tap - 3;
        if (jj >= 0) {
          const int io = dir ? (LSEQ - 1 - jj) : jj;
          acc += cw[dd * 4 + tap] * (float)s_xzh[dd * LSEQ + io];
        }
      }
      s_xc[dd * LSEQ + j] = siluf_(acc + cb[dd]);
    }
    __syncthreads();

    // x_proj: [18][128] = x_proj_w * xc ; rows 0-1 -> dt, 2-9 -> B, 10-17 -> C
    for (int idx = t; idx < 18 * LSEQ; idx += 256) {
      const int e = idx >> 7, j = idx & 127;
      float acc = 0.0f;
#pragma unroll
      for (int dd = 0; dd < DM; ++dd) acc += s_xc[dd * LSEQ + j] * xpw[e * DM + dd];
      if (e < 2)        s_dt[e * LSEQ + j] = acc;
      else if (e < 10)  s_B[(e - 2) * LSEQ + j] = acc;
      else              s_C[(e - 10) * LSEQ + j] = acc;
    }
    __syncthreads();

    // selective scan: thread = (dd, nn); 8-lane shuffle reduce over state dim
    {
      const int dd = t >> 3, nn = t & 7;
      const float w0 = dpw[dd * 2], w1 = dpw[dd * 2 + 1], b0 = dpb[dd];
      const float Av = -__expf(alg[dd * DSN + nn]);
      const float Dv = Dsk[dd];
      float h = 0.0f;
      for (int j = 0; j < LSEQ; ++j) {
        const float delta = softplusf_(b0 + w0 * s_dt[j] + w1 * s_dt[LSEQ + j]);
        const float xcv = s_xc[dd * LSEQ + j];
        h = __expf(delta * Av) * h + delta * s_B[nn * LSEQ + j] * xcv;
        float part = h * s_C[nn * LSEQ + j];
        part += __shfl_xor(part, 1);
        part += __shfl_xor(part, 2);
        part += __shfl_xor(part, 4);
        if (nn == 0) {
          const int io = dir ? (LSEQ - 1 - j) : j;
          const float zv = (float)s_xzh[(DM + dd) * LSEQ + io];
          s_yacc[dd * LSEQ + io] += 0.5f * (part + Dv * xcv) * siluf_(zv);
        }
      }
    }
    __syncthreads();
  }

  // ---- Phase 4: RMS norm -> f16, load out_proj weights ----
  for (int idx = t; idx < DM * DM; idx += 256) s_wo[idx] = (_Float16)opw[idx];
  if (t < LSEQ) {
    const int l = t;
    float ms = 0.0f;
#pragma unroll
    for (int dd = 0; dd < DM; ++dd) { const float v = s_yacc[dd * LSEQ + l]; ms += v * v; }
    const float sc = rsqrtf(ms * (1.0f / DM) + 1e-5f);
#pragma unroll
    for (int dd = 0; dd < DM; ++dd)
      s_o16[l * DM + dd] = (_Float16)(s_yacc[dd * LSEQ + l] * sc * rmsw[dd]);
  }
  __syncthreads();

  // ---- Phase 5: out_proj GEMM (128x32)x(32x32) via WMMA ----
  {
    const int m0 = wv * 16;
    v16h a = load_a16(s_o16, m0, DM, lane);
#pragma unroll
    for (int nt = 0; nt < 2; ++nt) {
      v16h bf = load_b16(s_wo, nt * 16, DM, lane);
      v8f c = {};
      c = __builtin_amdgcn_wmma_f32_16x16x32_f16(false, a, false, bf, (short)0, c, false, false);
      const int n  = nt * 16 + (lane & 15);
      const int hf = lane >> 4;
#pragma unroll
      for (int j = 0; j < 8; ++j)
        s_proj[(m0 + j + 8 * hf) * DM + n] = c[j];
    }
  }
  __syncthreads();

  // ---- Phase 6: logits + sigmoid ----
  if (t < LSEQ) {
    float acc = w2b[0];
#pragma unroll
    for (int o = 0; o < DM; ++o) acc += s_proj[t * DM + o] * w2w[o];
    out[bid * LSEQ + t] = sigmoidf_(acc);
  }
}

// ---------------------------------------------------------------------------
extern "C" void kernel_launch(void* const* d_in, const int* in_sizes, int n_in,
                              void* d_out, int out_size, void* d_ws, size_t ws_size,
                              hipStream_t stream) {
  (void)in_sizes; (void)n_in; (void)out_size; (void)ws_size;
  const float* x     = (const float*)d_in[0];
  const float* w1w   = (const float*)d_in[1];
  const float* w1b   = (const float*)d_in[2];
  const float* lng   = (const float*)d_in[3];
  const float* lnb   = (const float*)d_in[4];
  const float* ipw   = (const float*)d_in[5];
  const float* cwf   = (const float*)d_in[6];
  const float* cbf   = (const float*)d_in[7];
  const float* cwb   = (const float*)d_in[8];
  const float* cbb   = (const float*)d_in[9];
  const float* xpwf  = (const float*)d_in[10];
  const float* xpwb  = (const float*)d_in[11];
  const float* dpwf  = (const float*)d_in[12];
  const float* dpbf  = (const float*)d_in[13];
  const float* dpwb  = (const float*)d_in[14];
  const float* dpbb  = (const float*)d_in[15];
  const float* alogf = (const float*)d_in[16];
  const float* alogb = (const float*)d_in[17];
  const float* Dskf  = (const float*)d_in[18];
  const float* Dskb  = (const float*)d_in[19];
  const float* rmsw  = (const float*)d_in[20];
  const float* opw   = (const float*)d_in[21];
  const float* w2w   = (const float*)d_in[22];
  const float* w2b   = (const float*)d_in[23];

  float* wsf      = (float*)d_ws;
  float* partials = wsf;                     // 8192 floats
  float* pooled   = wsf + 8192;              // 2048 floats

  pool_partial<<<NBATCH * LSEQ * 4, 256, 0, stream>>>(x, partials);
  pool_final<<<8, 256, 0, stream>>>(partials, pooled);
  mamba_fused<<<NBATCH, 256, 0, stream>>>(pooled, w1w, w1b, lng, lnb, ipw,
                                          cwf, cbf, cwb, cbb, xpwf, xpwb,
                                          dpwf, dpbf, dpwb, dpbb, alogf, alogb,
                                          Dskf, Dskb, rmsw, opw, w2w, w2b,
                                          (float*)d_out);
}
